// Categorical2DSemanticMapModule_31430570672886
// MI455X (gfx1250) — matrix operand order repaired and checked by reading, not compiled, not run on hardware
//
#include <hip/hip_runtime.h>

// ---------------- constants (from reference) ----------------
#define BN   4      // batch
#define TN   4      // timesteps
#define CSEM 16     // semantic channels
#define NCH  22     // NC + NUM_SEM map channels
#define NCC  6      // NC
#define MLD  240    // local map size
#define MGD  960    // global map size
#define VRD  100    // vision range cells
#define ZBD  80     // z bins
#define MINH 13     // MIN_MAP_H
#define MAXH 25     // MAX_MAP_H
#define HS   120    // FRAME_H/DU
#define WSX  160    // FRAME_W/DU
#define PN   (HS*WSX)
#define DEG2RAD 0.017453292519943295f
#define RAD2DEG 57.29577951308232f

typedef __attribute__((ext_vector_type(16))) _Float16 v16h;
typedef __attribute__((ext_vector_type(8)))  float    v8f;

// ---------------- 4x4 semantic mean pooling via WMMA ----------------
// out(16ch x 16cols) = A(16x64 row-sums) x P(64x16 selection*1/16)
// Tile is staged through LDS with coalesced 128B wave loads, fragments are
// built from LDS (ds path), then two accumulating v_wmma_f32_16x16x32_f16.
#define CH_STRIDE 257   // 4*64 + 1 pad float -> spreads LDS banks across channels
__global__ void k_pool_sem(const float* __restrict__ obs, float* __restrict__ pool, int t) {
  __shared__ float tile[CSEM * CH_STRIDE];   // [ch][row0..3][col0..63], padded

  int wave = blockIdx.x;               // BN*HS*10 waves
  int b  = wave / (HS * 10);
  int r  = wave % (HS * 10);
  int ho = r / 10;                     // output row 0..119
  int jb = r % 10;                     // 16-output-col tile -> 64 input cols
  int lane = threadIdx.x;              // 0..31, wave32, EXEC all ones
  int mn   = lane & 15;                // A: M=channel ; B/D: N=out col
  int hi   = lane >> 4;

  // ---- coalesced stage: 4096 floats, 32 lanes x 128 iterations, 128B/instr ----
  const size_t obs_base = (((size_t)(b*TN + t))*20 + 4)*480*640;  // sem channel 0
#pragma unroll 8
  for (int n = lane; n < CSEM*4*64; n += 32) {
    int ch  = n >> 8;          // /256
    int row = (n >> 6) & 3;
    int col = n & 63;
    tile[ch*CH_STRIDE + row*64 + col] =
        obs[obs_base + ((size_t)ch*480 + (ho*4 + row))*640 + jb*64 + col];
  }
  __syncthreads();

  const float* tch = &tile[mn * CH_STRIDE];  // A fragment: channel = mn

  v8f acc = {};
#pragma unroll
  for (int chunk = 0; chunk < 2; ++chunk) {
    int xbase = chunk * 32;
    v16h a, bm;
#pragma unroll
    for (int h = 0; h < 16; ++h) {
      // A layout: halves 0..7 -> K = 8*hi + h ; halves 8..15 -> K = 16 + 8*hi + (h-8)
      int kA = (h < 8) ? (hi*8 + h) : (16 + hi*8 + (h - 8));
      int x = xbase + kA;
      float s = tch[x] + tch[64 + x] + tch[128 + x] + tch[192 + x];
      a[h] = (_Float16)s;              // integer 0..4, exact in f16
      // B layout: halves 0..7 -> K = 8*hi + h ; halves 8..15 -> K = 16 + 8*hi + (h-8) ; value = 1/16 if input block matches out col
      int kB = (h < 8) ? (hi*8 + h) : (16 + hi*8 + (h - 8));
      int xb = xbase + kB;
      bm[h] = (_Float16)(((xb >> 2) == mn) ? 0.0625f : 0.0f);
    }
    acc = __builtin_amdgcn_wmma_f32_16x16x32_f16(false, a, false, bm,
                                                 (short)0, acc, false, false);
  }
  // D layout: VGPR r -> M = r + 8*hi (channel), N = mn (out col)
#pragma unroll
  for (int rr = 0; rr < 8; ++rr) {
    int ch = rr + hi*8;
    int wo = jb*16 + mn;
    pool[(((size_t)b*CSEM + ch)*HS + ho)*WSX + wo] = acc[rr];
  }
}

// ---------------- pose update ----------------
__global__ void k_pose(float* pose, const float* pd, const int* dones, int t) {
  int b = threadIdx.x;
  if (b >= BN) return;
  float px = pose[b*3], py = pose[b*3+1], pt = pose[b*3+2];
  if (dones[b*TN + t]) { px = 6.f; py = 6.f; pt = 0.f; }
  const float* q = pd + (size_t)(b*TN + t)*3;
  float tr = pt * DEG2RAD;
  float cn = cosf(tr), sn = sinf(tr);
  float nx = px + q[0]*cn - q[1]*sn;
  float ny = py + q[0]*sn + q[1]*cn;
  float nt = fmodf(pt + q[2]*RAD2DEG + 180.f, 360.f);
  if (nt < 0.f) nt += 360.f;
  nt -= 180.f;
  pose[b*3] = nx; pose[b*3+1] = ny; pose[b*3+2] = nt;
}

// ---------------- conditional map reset on done (b128 path) ----------------
__global__ void k_reset4(float4* __restrict__ m, const int* __restrict__ dones,
                         int t, size_t per_b4, size_t total4) {
  size_t idx = (size_t)blockIdx.x*blockDim.x + threadIdx.x;
  if (idx >= total4) return;
  int b = (int)(idx / per_b4);
  if (dones[b*TN + t]) m[idx] = make_float4(0.f, 0.f, 0.f, 0.f);
}

// ---------------- voxel splat collapsed over z ----------------
__global__ void k_splat(const float* __restrict__ obs, const float* __restrict__ cam,
                        const float* __restrict__ pool,
                        float* __restrict__ agent, float* __restrict__ expg, int t) {
  int idx = blockIdx.x*blockDim.x + threadIdx.x;
  if (idx >= BN*PN) return;
  int b = idx / PN;
  int p = idx % PN;
  int i = p / WSX, j = p % WSX;
  float d = obs[((((size_t)(b*TN + t))*20 + 3)*480 + (size_t)i*4)*640 + (size_t)j*4];
  if (!(d >= 50.f && d <= 350.f)) return;
  float focal = 640.0f / (2.0f * tanf(39.5f * DEG2RAD));
  float xs = ((float)(j*4) - 320.0f) / focal;
  float zr = (240.0f - (float)(i*4)) / focal;
  float elev = atan2f(cam[b*16 + 9], cam[b*16 + 10]);
  float ce = cosf(elev), se = sinf(elev);
  float lat = d * xs;
  float up  = d * zr;
  float fwd = d*ce + up*se;
  float hgt = -d*se + up*ce + 88.0f;
  float gx = lat*0.2f + 50.0f;
  float gy = fwd*0.2f;
  float gz = hgt*0.2f + 8.0f;
  float x0 = floorf(gx), y0 = floorf(gy), z0 = floorf(gz);

  float feat[CSEM];
#pragma unroll
  for (int c = 0; c < CSEM; ++c)
    feat[c] = pool[(((size_t)b*CSEM + c)*HS + i)*WSX + j];

  for (int dx = 0; dx <= 1; ++dx)
  for (int dy = 0; dy <= 1; ++dy) {
    float xi = x0 + dx, yi = y0 + dy;
    if (xi < 0.f || xi >= (float)VRD || yi < 0.f || yi >= (float)VRD) continue;
    float wxy = (1.f - fabsf(gx - xi)) * (1.f - fabsf(gy - yi));
    float wAll = 0.f, wAg = 0.f;
    for (int dz = 0; dz <= 1; ++dz) {
      float zi = z0 + dz;
      if (zi < 0.f || zi >= (float)ZBD) continue;
      float wz = 1.f - fabsf(gz - zi);
      wAll += wz;
      int zii = (int)zi;
      if (zii >= MINH && zii < MAXH) wAg += wz;
    }
    int cell = (int)yi * VRD + (int)xi;
    float we = wxy * wAll;
    if (we != 0.f) atomicAdd(&expg[b*VRD*VRD + cell], we);
    float wa = wxy * wAg;
    if (wa != 0.f) {
      atomicAdd(&agent[(size_t)b*17*VRD*VRD + cell], wa);
#pragma unroll
      for (int c = 0; c < CSEM; ++c) {
        float v = feat[c] * wa;
        if (v != 0.f) atomicAdd(&agent[((size_t)b*17 + 1 + c)*VRD*VRD + cell], v);
      }
    }
  }
}

// ---------------- agent view window fill ----------------
__global__ void k_agent_view(const float* __restrict__ agent, const float* __restrict__ expg,
                             float* __restrict__ av) {
  int idx = blockIdx.x*blockDim.x + threadIdx.x;
  if (idx >= BN*VRD*VRD) return;
  int b = idx / (VRD*VRD);
  int cell = idx % (VRD*VRD);
  int yy = cell / VRD, xx = cell % VRD;
  int h = 120 + yy, w = 70 + xx;  // y1=ML/2, x1=ML/2-VR/2
  const size_t plane = (size_t)MLD*MLD;
  size_t po = ((size_t)b*NCH*MLD + h)*MLD + w;
  float a0 = agent[(size_t)b*17*VRD*VRD + cell];
  av[po]          = fminf(fmaxf(a0, 0.f), 1.f);                     // fp_map (MAP_THR=1)
  av[po + plane]  = fminf(fmaxf(expg[b*VRD*VRD + cell], 0.f), 1.f); // fp_exp (EXP_THR=1)
#pragma unroll
  for (int c = 0; c < CSEM; ++c) {
    float s = agent[((size_t)b*17 + 1 + c)*VRD*VRD + cell];
    av[po + (size_t)(NCC + c)*plane] = fminf(fmaxf(s*0.2f, 0.f), 1.f); // CAT_THR=5
  }
}

// ---------------- bilinear sampler (zero padding, clipped gather) ----------------
__device__ __forceinline__ void sample22(const float* __restrict__ src, float* __restrict__ dst,
                                         int b, int h, int w, float gxn, float gyn) {
  float x = (gxn + 1.f) * 0.5f * (float)(MLD - 1);
  float y = (gyn + 1.f) * 0.5f * (float)(MLD - 1);
  float x0f = floorf(x), y0f = floorf(y);
  size_t mapb = (size_t)b*NCH*MLD*MLD;
  size_t opix = (size_t)h*MLD + w;
  int   idxs[4];
  float wgt[4];
  int n = 0;
  for (int dy = 0; dy <= 1; ++dy)
  for (int dx = 0; dx <= 1; ++dx) {
    float xi = x0f + dx, yi = y0f + dy;
    bool ok = (xi >= 0.f) && (xi <= (float)(MLD-1)) && (yi >= 0.f) && (yi <= (float)(MLD-1));
    float wv = (1.f - fabsf(x - xi)) * (1.f - fabsf(y - yi));
    float xc = fminf(fmaxf(xi, 0.f), (float)(MLD-1));
    float yc = fminf(fmaxf(yi, 0.f), (float)(MLD-1));
    idxs[n] = (int)yc*MLD + (int)xc;
    wgt[n]  = ok ? wv : 0.f;
    ++n;
  }
#pragma unroll
  for (int c = 0; c < NCH; ++c) {
    const float* s = src + mapb + (size_t)c*MLD*MLD;
    dst[mapb + (size_t)c*MLD*MLD + opix] =
        s[idxs[0]]*wgt[0] + s[idxs[1]]*wgt[1] + s[idxs[2]]*wgt[2] + s[idxs[3]]*wgt[3];
  }
}

__global__ void k_warp_rot(const float* __restrict__ src, float* __restrict__ dst,
                           const float* __restrict__ pose) {
  int idx = blockIdx.x*blockDim.x + threadIdx.x;
  if (idx >= BN*MLD*MLD) return;
  int b = idx / (MLD*MLD);
  int r = idx % (MLD*MLD);
  int h = r / MLD, w = r % MLD;
  float th = (90.f - pose[b*3+2]) * DEG2RAD;
  float cn = cosf(th), sn = sinf(th);
  float xg = -1.f + 2.f*(float)w/(float)(MLD-1);
  float yg = -1.f + 2.f*(float)h/(float)(MLD-1);
  sample22(src, dst, b, h, w, cn*xg - sn*yg, sn*xg + cn*yg);
}

__global__ void k_warp_trans(const float* __restrict__ src, float* __restrict__ dst,
                             const float* __restrict__ pose) {
  int idx = blockIdx.x*blockDim.x + threadIdx.x;
  if (idx >= BN*MLD*MLD) return;
  int b = idx / (MLD*MLD);
  int r = idx % (MLD*MLD);
  int h = r / MLD, w = r % MLD;
  float stx = -(pose[b*3+0]*20.f - 120.f) / 120.f;
  float sty = -(pose[b*3+1]*20.f - 120.f) / 120.f;
  float xg = -1.f + 2.f*(float)w/(float)(MLD-1);
  float yg = -1.f + 2.f*(float)h/(float)(MLD-1);
  sample22(src, dst, b, h, w, xg + stx, yg + sty);
}

// ---------------- fuse warped view into local map ----------------
__global__ void k_fuse(const float* __restrict__ w2, float* __restrict__ fl,
                       const float* __restrict__ pose) {
  int idx = blockIdx.x*blockDim.x + threadIdx.x;
  if (idx >= BN*MLD*MLD) return;
  int b = idx / (MLD*MLD);
  int r = idx % (MLD*MLD);
  int h = r / MLD, w = r % MLD;
  size_t mapb = (size_t)b*NCH*MLD*MLD;
  size_t pix  = (size_t)h*MLD + w;
  // 3x3 max pool (SAME, -inf pad) on channel 0
  float m = -1e30f;
  for (int dh = -1; dh <= 1; ++dh) {
    int hh = h + dh; if (hh < 0 || hh >= MLD) continue;
    for (int dw = -1; dw <= 1; ++dw) {
      int ww = w + dw; if (ww < 0 || ww >= MLD) continue;
      m = fmaxf(m, w2[mapb + (size_t)hh*MLD + ww]);
    }
  }
  fl[mapb + pix] = fmaxf(fl[mapb + pix], m);

  float ddx = (float)w - pose[b*3+0]*20.f;   // 100/RES = 20
  float ddy = (float)h - pose[b*3+1]*20.f;
  float d2 = ddx*ddx + ddy*ddy;
  float curr = (d2 <= 4.f)    ? 1.f : 0.f;
  float bc   = (d2 <= 1600.f) ? 1.f : 0.f;   // BC_R^2
#pragma unroll
  for (int c = 1; c < NCH; ++c) {
    size_t o = mapb + (size_t)c*MLD*MLD + pix;
    float v = fmaxf(fl[o], w2[o]);
    if (c == 2)      v = curr;
    else if (c == 3) v = fmaxf(v, curr);
    else if (c == 4) v = fmaxf(v, bc);
    fl[o] = v;
  }
}

// ---------------- conditional global map slice update (b128 path) ----------------
__global__ void k_upd_global(const float* __restrict__ fl, float* __restrict__ fg,
                             const int* __restrict__ updv, const int* __restrict__ lmb, int t) {
  const int W4 = MLD/4;
  int idx = blockIdx.x*blockDim.x + threadIdx.x;
  if (idx >= BN*NCH*MLD*W4) return;
  int b = idx / (NCH*MLD*W4);
  if (!updv[b*TN + t]) return;
  int r  = idx % (NCH*MLD*W4);
  int c  = r / (MLD*W4);
  int rr = r % (MLD*W4);
  int h = rr / W4, w4 = rr % W4;
  float4 val = *(const float4*)(fl + (size_t)idx*4);   // fl enumerated in same order
  int r0 = lmb[b*4 + 0], c0 = lmb[b*4 + 2];
  size_t gofs = (((size_t)b*NCH + c)*MGD + (r0 + h))*MGD + c0 + w4*4;
  if ((c0 & 3) == 0) {
    *(float4*)(fg + gofs) = val;
  } else {
    fg[gofs]   = val.x; fg[gofs+1] = val.y;
    fg[gofs+2] = val.z; fg[gofs+3] = val.w;
  }
}

// ---------------- map features output (local | gmax | local-sem), b128 ----------------
__global__ void k_mf(const float* __restrict__ fl, const float* __restrict__ fg,
                     float* __restrict__ omf, int t) {
  const int W4 = MLD/4;
  int idx = blockIdx.x*blockDim.x + threadIdx.x;
  if (idx >= BN*28*MLD*W4) return;
  int b  = idx / (28*MLD*W4);
  int r  = idx % (28*MLD*W4);
  int c  = r / (MLD*W4);
  int rr = r % (MLD*W4);
  int h = rr / W4, w4 = rr % W4;
  float4 v;
  if (c < NCC || c >= 2*NCC) {
    int lc = (c < NCC) ? c : (c - NCC);
    v = *(const float4*)(fl + (((size_t)b*NCH + lc)*MLD + h)*MLD + w4*4);
  } else {
    int ch = c - NCC;
    const float* g = fg + (((size_t)b*NCH + ch)*MGD + (size_t)h*4)*MGD + (size_t)w4*16;
    float m0 = -1e30f, m1 = -1e30f, m2 = -1e30f, m3 = -1e30f;
#pragma unroll
    for (int di = 0; di < 4; ++di) {
      const float4* grow = (const float4*)(g + (size_t)di*MGD);
      float4 q0 = grow[0], q1 = grow[1], q2 = grow[2], q3 = grow[3];
      m0 = fmaxf(m0, fmaxf(fmaxf(q0.x, q0.y), fmaxf(q0.z, q0.w)));
      m1 = fmaxf(m1, fmaxf(fmaxf(q1.x, q1.y), fmaxf(q1.z, q1.w)));
      m2 = fmaxf(m2, fmaxf(fmaxf(q2.x, q2.y), fmaxf(q2.z, q2.w)));
      m3 = fmaxf(m3, fmaxf(fmaxf(q3.x, q3.y), fmaxf(q3.z, q3.w)));
    }
    v = make_float4(m0, m1, m2, m3);
  }
  *(float4*)(omf + ((((size_t)(b*TN + t))*28 + c)*MLD + h)*MLD + w4*4) = v;
}

// ---------------- pose outputs ----------------
__global__ void k_posesout(const float* pose, const float* origins,
                           float* lp, float* gp, int t) {
  int b = threadIdx.x;
  if (b >= BN) return;
#pragma unroll
  for (int k = 0; k < 3; ++k) {
    float p = pose[b*3 + k];
    lp[(size_t)(b*TN + t)*3 + k] = p;
    gp[(size_t)(b*TN + t)*3 + k] = p + origins[b*3 + k];
  }
}

__global__ void k_lmborg(const int* lmb, const float* org, float* olmb, float* oorg) {
  int i = threadIdx.x;
  if (i >= BN*TN) return;
  int b = i / TN;
#pragma unroll
  for (int k = 0; k < 4; ++k) olmb[i*4 + k] = (float)lmb[b*4 + k];
#pragma unroll
  for (int k = 0; k < 3; ++k) oorg[i*3 + k] = org[b*3 + k];
}

// ---------------- host-side launch ----------------
extern "C" void kernel_launch(void* const* d_in, const int* in_sizes, int n_in,
                              void* d_out, int out_size, void* d_ws, size_t ws_size,
                              hipStream_t stream) {
  const float* obs     = (const float*)d_in[0];
  const float* pd      = (const float*)d_in[1];
  const float* cam     = (const float*)d_in[2];
  const float* init_l  = (const float*)d_in[3];
  const float* init_g  = (const float*)d_in[4];
  const float* init_lp = (const float*)d_in[5];
  const float* origins = (const float*)d_in[7];
  const int*   dones   = (const int*)d_in[8];
  const int*   updv    = (const int*)d_in[9];
  const int*   lmb     = (const int*)d_in[10];

  const size_t N_MF = (size_t)BN*TN*28*MLD*MLD;   // 25,804,800
  const size_t N_FL = (size_t)BN*NCH*MLD*MLD;     //  5,068,800
  const size_t N_FG = (size_t)BN*NCH*MGD*MGD;     // 81,100,800

  float* out   = (float*)d_out;
  float* o_mf  = out;
  float* o_fl  = o_mf + N_MF;     // local map carry lives here
  float* o_fg  = o_fl + N_FL;     // global map carry lives here
  float* o_lp  = o_fg + N_FG;
  float* o_gp  = o_lp + (size_t)BN*TN*3;
  float* o_lmb = o_gp + (size_t)BN*TN*3;
  float* o_org = o_lmb + (size_t)BN*TN*4;

  float* ws      = (float*)d_ws;
  float* w_pool  = ws;                                      // BN*16*120*160
  float* w_agent = w_pool  + (size_t)BN*CSEM*HS*WSX;        // BN*17*100*100
  float* w_exp   = w_agent + (size_t)BN*17*VRD*VRD;         // BN*100*100
  float* w_av    = w_exp   + (size_t)BN*VRD*VRD;            // BN*22*240*240 (agent_view / warped2)
  float* w_w1    = w_av    + N_FL;                          // BN*22*240*240 (warped1)
  float* w_pose  = w_w1    + N_FL;                          // BN*3

  // init carries from inputs (deterministic across replays)
  hipMemcpyAsync(o_fl, init_l, N_FL*sizeof(float), hipMemcpyDeviceToDevice, stream);
  hipMemcpyAsync(o_fg, init_g, N_FG*sizeof(float), hipMemcpyDeviceToDevice, stream);
  hipMemcpyAsync(w_pose, init_lp, (size_t)BN*3*sizeof(float), hipMemcpyDeviceToDevice, stream);
  k_lmborg<<<1, 32, 0, stream>>>(lmb, origins, o_lmb, o_org);

  const int thr = 256;
  for (int t = 0; t < TN; ++t) {
    // done resets (float4 granularity; all sizes divisible by 4)
    k_reset4<<<(int)((N_FL/4 + thr - 1)/thr), thr, 0, stream>>>((float4*)o_fl, dones, t,
        (size_t)NCH*MLD*MLD/4, N_FL/4);
    k_reset4<<<(int)((N_FG/4 + thr - 1)/thr), thr, 0, stream>>>((float4*)o_fg, dones, t,
        (size_t)NCH*MGD*MGD/4, N_FG/4);
    k_pose<<<1, 32, 0, stream>>>(w_pose, pd, dones, t);

    // per-step scratch clears (agent+exp are contiguous; av reused as warped2)
    hipMemsetAsync(w_agent, 0, (size_t)(BN*17*VRD*VRD + BN*VRD*VRD)*sizeof(float), stream);
    hipMemsetAsync(w_av, 0, N_FL*sizeof(float), stream);

    // semantic 4x4 mean pooling via LDS-staged WMMA
    k_pool_sem<<<BN*HS*10, 32, 0, stream>>>(obs, w_pool, t);
    // depth splat collapsed over z
    k_splat<<<(BN*PN + thr - 1)/thr, thr, 0, stream>>>(obs, cam, w_pool, w_agent, w_exp, t);
    // thresholded agent view window
    k_agent_view<<<(BN*VRD*VRD + thr - 1)/thr, thr, 0, stream>>>(w_agent, w_exp, w_av);
    // rotation warp then translation warp (av -> w1 -> av)
    k_warp_rot  <<<(BN*MLD*MLD + thr - 1)/thr, thr, 0, stream>>>(w_av, w_w1, w_pose);
    k_warp_trans<<<(BN*MLD*MLD + thr - 1)/thr, thr, 0, stream>>>(w_w1, w_av, w_pose);
    // fuse into local map (3x3 maxpool on ch0, agent disks)
    k_fuse<<<(BN*MLD*MLD + thr - 1)/thr, thr, 0, stream>>>(w_av, o_fl, w_pose);
    // conditional global slice update (b128)
    k_upd_global<<<(BN*NCH*MLD*(MLD/4) + thr - 1)/thr, thr, 0, stream>>>(o_fl, o_fg, updv, lmb, t);
    // map features output (b128)
    k_mf<<<(BN*28*MLD*(MLD/4) + thr - 1)/thr, thr, 0, stream>>>(o_fl, o_fg, o_mf, t);
    // pose outputs
    k_posesout<<<1, 32, 0, stream>>>(w_pose, origins, o_lp, o_gp, t);
  }
}
